// PersistentGRUController_58454504898729
// MI455X (gfx1250) — matrix operand order repaired and checked by reading, not compiled, not run on hardware
//
#include <hip/hip_runtime.h>
#include <math.h>

#define NWG   32
#define TPB   256
#define NWAVE (NWG * (TPB / 32))
#define NTHR  (NWG * TPB)

typedef __bf16 bf16;
typedef __attribute__((ext_vector_type(8)))  __bf16 v8bf;
typedef __attribute__((ext_vector_type(16))) __bf16 v16bf;
typedef __attribute__((ext_vector_type(8)))  float  v8f;

constexpr int Lc = 2, Bc = 64, Tc = 512, Ic = 512, Hc = 512, Pc = 256, Oc = 512;
constexpr int IH = Ic + Hc; // 1024

struct GruParams {
  const float *x, *Wr, *br, *Wu, *bu, *Wn, *bn;
  const float *Wmr, *bmr, *Wmw, *bmw, *Wmu, *bmu, *Wmh, *bmh;
  const float *Wd, *bd, *Wsr, *bs, *Wout, *bout, *dopap, *serop;
  float *out;
  unsigned *bar;
  bf16 *wr, *wu, *wn, *wmr, *wmw, *wmu, *wd, *wsr, *wmh, *wout;
  bf16 *xbf, *hbf, *rhbf, *hcbf, *mobf;
  float *h32, *m32, *u32, *hc32, *et32;
};

__device__ inline float sigf(float v) { return 1.0f / (1.0f + expf(-v)); }

__device__ inline void grid_sync(unsigned *bar) {
  __threadfence();
  __syncthreads();
  if (threadIdx.x == 0) {
    unsigned gen  = __hip_atomic_load(&bar[1], __ATOMIC_RELAXED, __HIP_MEMORY_SCOPE_AGENT);
    unsigned prev = __hip_atomic_fetch_add(&bar[0], 1u, __ATOMIC_ACQ_REL, __HIP_MEMORY_SCOPE_AGENT);
    if (prev == NWG - 1) {
      __hip_atomic_store(&bar[0], 0u, __ATOMIC_RELAXED, __HIP_MEMORY_SCOPE_AGENT);
      __hip_atomic_fetch_add(&bar[1], 1u, __ATOMIC_RELEASE, __HIP_MEMORY_SCOPE_AGENT);
    } else {
      while (__hip_atomic_load(&bar[1], __ATOMIC_ACQUIRE, __HIP_MEMORY_SCOPE_AGENT) == gen)
        __builtin_amdgcn_s_sleep(1);
    }
  }
  __syncthreads();
  __threadfence();
}

// C(16x16,f32) += A(16x32,bf16) x B(32x16,bf16), B = W^T with W [N,K] row-major.
// A columns [0,512) come from A0, [512,K) from A1 (both ld = ldA).
__device__ inline v8f mm_tile(const bf16 *__restrict__ A0, const bf16 *__restrict__ A1,
                              int ldA, const bf16 *__restrict__ W, int ldW,
                              int m0, int n0, int K, int lane) {
  const int lm = lane & 15;
  const int hi = lane >> 4;
  v8f acc = {};
  const int aoff = (m0 + lm) * ldA + hi * 8;
  const bf16 *wb = W + (size_t)(n0 + lm) * ldW + hi * 8;
  for (int k0 = 0; k0 < K; k0 += 32) {
    const bf16 *ab = (k0 < Ic) ? (A0 + aoff + k0) : (A1 + aoff + (k0 - Ic));
    v8bf al = *(const v8bf *)(ab);
    v8bf ah = *(const v8bf *)(ab + 16);
    v8bf bl = *(const v8bf *)(wb + k0);
    v8bf bh = *(const v8bf *)(wb + k0 + 16);
    v16bf av = __builtin_shufflevector(al, ah, 0,1,2,3,4,5,6,7,8,9,10,11,12,13,14,15);
    v16bf bv = __builtin_shufflevector(bl, bh, 0,1,2,3,4,5,6,7,8,9,10,11,12,13,14,15);
    acc = __builtin_amdgcn_wmma_f32_16x16x32_bf16(false, av, false, bv, (short)0, acc, false, false);
  }
  return acc;
}

// ---- stage A: fused reset/update gates. r -> rhbf (r*h), u -> u32 ----
__device__ inline void stageA(const GruParams &p, int l, const bf16 *inp, int gwave, int lane) {
  const bf16 *A1 = p.hbf + (size_t)l * Bc * Hc;
  for (int tile = gwave; tile < 4 * 64; tile += NWAVE) {
    int mt = tile >> 6, nt = tile & 63;
    int m0 = mt * 16, n0g = nt * 16;
    bool isr = (n0g < Hc);
    int n0 = isr ? n0g : n0g - Hc;
    const bf16 *W = (isr ? p.wr : p.wu) + (size_t)l * Hc * IH;
    v8f acc = mm_tile(inp, A1, Hc, W, IH, m0, n0, IH, lane);
    int lm = lane & 15, hi = lane >> 4;
    int n = n0 + lm;
    if (isr) {
      float bias = p.br[l * Hc + n];
      for (int i = 0; i < 8; ++i) {
        int m = m0 + hi * 8 + i;
        float r = sigf(acc[i] + bias);
        float hv = p.h32[((size_t)l * Bc + m) * Hc + n];
        p.rhbf[m * Hc + n] = (bf16)(r * hv);
      }
    } else {
      float bias = p.bu[l * Hc + n];
      for (int i = 0; i < 8; ++i) {
        int m = m0 + hi * 8 + i;
        p.u32[m * Hc + n] = sigf(acc[i] + bias);
      }
    }
  }
}

// ---- stage B: candidate + h_cand ----
__device__ inline void stageB(const GruParams &p, int l, const bf16 *inp, int gwave, int lane) {
  const bf16 *W = p.wn + (size_t)l * Hc * IH;
  for (int tile = gwave; tile < 4 * 32; tile += NWAVE) {
    int mt = tile >> 5, nt = tile & 31;
    int m0 = mt * 16, n0 = nt * 16;
    v8f acc = mm_tile(inp, p.rhbf, Hc, W, IH, m0, n0, IH, lane);
    int lm = lane & 15, hi = lane >> 4;
    int n = n0 + lm;
    float bias = p.bn[l * Hc + n];
    for (int i = 0; i < 8; ++i) {
      int m = m0 + hi * 8 + i;
      float nh = tanhf(acc[i] + bias);
      float u = p.u32[m * Hc + n];
      float hv = p.h32[((size_t)l * Bc + m) * Hc + n];
      float hc = (1.0f - u) * hv + u * nh;
      p.hc32[m * Hc + n] = hc;
      p.hcbf[m * Hc + n] = (bf16)hc;
    }
  }
}

// ---- stage C: read gate * m -> mobf ----
__device__ inline void stageC(const GruParams &p, int l, int gwave, int lane) {
  const bf16 *W = p.wmr + (size_t)l * Pc * Hc;
  for (int tile = gwave; tile < 4 * 16; tile += NWAVE) {
    int mt = tile >> 4, nt = tile & 15;
    int m0 = mt * 16, n0 = nt * 16;
    v8f acc = mm_tile(p.hcbf, p.hcbf, Hc, W, Hc, m0, n0, Hc, lane);
    int lm = lane & 15, hi = lane >> 4;
    int n = n0 + lm;
    float bias = p.bmr[l * Pc + n];
    for (int i = 0; i < 8; ++i) {
      int m = m0 + hi * 8 + i;
      float rg = sigf(acc[i] + bias);
      float mv = p.m32[((size_t)l * Bc + m) * Pc + n];
      p.mobf[m * Pc + n] = (bf16)(rg * mv);
    }
  }
}

// ---- stage D: h_mem = h_cand + mem_out @ Wmh^T + bmh; becomes new h ----
__device__ inline void stageD(const GruParams &p, int l, int gwave, int lane) {
  const bf16 *W = p.wmh + (size_t)l * Hc * Pc;
  for (int tile = gwave; tile < 4 * 32; tile += NWAVE) {
    int mt = tile >> 5, nt = tile & 31;
    int m0 = mt * 16, n0 = nt * 16;
    v8f acc = mm_tile(p.mobf, p.mobf, Pc, W, Pc, m0, n0, Pc, lane);
    int lm = lane & 15, hi = lane >> 4;
    int n = n0 + lm;
    float bias = p.bmh[l * Hc + n];
    for (int i = 0; i < 8; ++i) {
      int m = m0 + hi * 8 + i;
      float hm = p.hc32[m * Hc + n] + acc[i] + bias;
      p.h32[((size_t)l * Bc + m) * Hc + n] = hm;
      p.hbf[((size_t)l * Bc + m) * Hc + n] = (bf16)hm;
    }
  }
}

// ---- stage E1: fused Wmw/Wmu/Wd/Ws GEMM (raw pre-activations into et32) ----
__device__ inline void stageE1(const GruParams &p, int l, int gwave, int lane, float dopa, float sero) {
  const bf16 *A = p.hbf + (size_t)l * Bc * Hc;
  const size_t lPH = (size_t)l * Pc * Hc;
  const bf16 *wq[4]  = { p.wmw + lPH, p.wmu + lPH, p.wd + lPH, p.wsr + lPH };
  const float *bq[4] = { p.bmw + (size_t)l * Pc, p.bmu + (size_t)l * Pc,
                         p.bd  + (size_t)l * Pc, p.bs  + (size_t)l * Pc };
  for (int tile = gwave; tile < 4 * 64; tile += NWAVE) {
    int mt = tile >> 6, nt = tile & 63;
    int m0 = mt * 16, n0g = nt * 16;
    int q = n0g >> 8, n0 = n0g & 255;
    v8f acc = mm_tile(A, A, Hc, wq[q], Hc, m0, n0, Hc, lane);
    float scale = (q == 2) ? dopa : (q == 3) ? sero : 1.0f;
    int lm = lane & 15, hi = lane >> 4;
    int n = n0 + lm;
    float bias = bq[q][n];
    for (int i = 0; i < 8; ++i) {
      int m = m0 + hi * 8 + i;
      p.et32[m * 1024 + q * Pc + n] = (acc[i] + bias) * scale;
    }
  }
}

// ---- stage E2: memory state update ----
__device__ inline void stageE2(const GruParams &p, int l, int gtid) {
  for (int idx = gtid; idx < Bc * Pc; idx += NTHR) {
    int m = idx >> 8, n = idx & 255;
    const float *e = p.et32 + m * 1024;
    float wg  = sigf(e[n]) * sigf(e[2 * Pc + n]) * (1.0f - sigf(e[3 * Pc + n]));
    float nmc = tanhf(e[Pc + n]);
    float *mp = p.m32 + ((size_t)l * Bc + m) * Pc + n;
    float mv = *mp;
    *mp = (1.0f - wg) * mv + wg * nmc;
  }
}

// ---- output head ----
__device__ inline void stageOUT(const GruParams &p, int t, int gwave, int lane) {
  const bf16 *A = p.hbf + (size_t)(Lc - 1) * Bc * Hc;
  for (int tile = gwave; tile < 4 * 32; tile += NWAVE) {
    int mt = tile >> 5, nt = tile & 31;
    int m0 = mt * 16, n0 = nt * 16;
    v8f acc = mm_tile(A, A, Hc, p.wout, Hc, m0, n0, Hc, lane);
    int lm = lane & 15, hi = lane >> 4;
    int n = n0 + lm;
    float bias = p.bout[n];
    for (int i = 0; i < 8; ++i) {
      int m = m0 + hi * 8 + i;
      p.out[((size_t)m * Tc + t) * Oc + n] = acc[i] + bias;
    }
  }
}

__device__ inline void xconv(const GruParams &p, int t, int gtid) {
  for (int idx = gtid; idx < Bc * Ic; idx += NTHR) {
    int b = idx >> 9, i = idx & 511;
    p.xbf[idx] = (bf16)p.x[((size_t)b * Tc + t) * Ic + i];
  }
}

__device__ inline void cvt(const float *s, bf16 *d, int n, int gtid) {
  for (int i = gtid; i < n; i += NTHR) d[i] = (bf16)s[i];
}

__global__ __launch_bounds__(TPB) void gru_persistent(GruParams p) {
  const int lane  = threadIdx.x & 31;
  const int gwave = blockIdx.x * (TPB / 32) + (threadIdx.x >> 5);
  const int gtid  = blockIdx.x * TPB + threadIdx.x;
  const float dopa = p.dopap[0];
  const float sero = p.serop[0];

  // Prologue: weights fp32->bf16, zero recurrent state, stage x(t=0)
  cvt(p.Wr,   p.wr,   Lc * Hc * IH, gtid);
  cvt(p.Wu,   p.wu,   Lc * Hc * IH, gtid);
  cvt(p.Wn,   p.wn,   Lc * Hc * IH, gtid);
  cvt(p.Wmr,  p.wmr,  Lc * Pc * Hc, gtid);
  cvt(p.Wmw,  p.wmw,  Lc * Pc * Hc, gtid);
  cvt(p.Wmu,  p.wmu,  Lc * Pc * Hc, gtid);
  cvt(p.Wd,   p.wd,   Lc * Pc * Hc, gtid);
  cvt(p.Wsr,  p.wsr,  Lc * Pc * Hc, gtid);
  cvt(p.Wmh,  p.wmh,  Lc * Hc * Pc, gtid);
  cvt(p.Wout, p.wout, Oc * Hc,      gtid);
  for (int i = gtid; i < Lc * Bc * Hc; i += NTHR) { p.h32[i] = 0.0f; p.hbf[i] = (bf16)0.0f; }
  for (int i = gtid; i < Lc * Bc * Pc; i += NTHR) p.m32[i] = 0.0f;
  xconv(p, 0, gtid);
  grid_sync(p.bar);

  for (int t = 0; t < Tc; ++t) {
    if (t > 0) stageE2(p, 1, gtid);      // finish previous step's layer-1 memory
    stageA(p, 0, p.xbf, gwave, lane);    grid_sync(p.bar);
    stageB(p, 0, p.xbf, gwave, lane);    grid_sync(p.bar);
    stageC(p, 0, gwave, lane);           grid_sync(p.bar);
    stageD(p, 0, gwave, lane);           grid_sync(p.bar);
    stageA(p, 1, p.hbf, gwave, lane);    // inp for layer1 = new h_mem of layer0
    stageE1(p, 0, gwave, lane, dopa, sero);
    grid_sync(p.bar);
    stageB(p, 1, p.hbf, gwave, lane);
    stageE2(p, 0, gtid);
    grid_sync(p.bar);
    stageC(p, 1, gwave, lane);           grid_sync(p.bar);
    stageD(p, 1, gwave, lane);           grid_sync(p.bar);
    stageOUT(p, t, gwave, lane);
    stageE1(p, 1, gwave, lane, dopa, sero);
    if (t + 1 < Tc) xconv(p, t + 1, gtid);
    grid_sync(p.bar);
  }
  stageE2(p, 1, gtid);                   // last step's layer-1 memory update
  grid_sync(p.bar);

  const size_t OB = (size_t)Bc * Tc * Oc;
  for (int i = gtid; i < Lc * Bc * Hc; i += NTHR) p.out[OB + i] = p.h32[i];
  for (int i = gtid; i < Lc * Bc * Pc; i += NTHR) p.out[OB + Lc * Bc * Hc + i] = p.m32[i];
}

extern "C" void kernel_launch(void *const *d_in, const int *in_sizes, int n_in,
                              void *d_out, int out_size, void *d_ws, size_t ws_size,
                              hipStream_t stream) {
  (void)in_sizes; (void)n_in; (void)out_size; (void)ws_size;
  GruParams p;
  p.x    = (const float *)d_in[0];
  p.Wr   = (const float *)d_in[1];   p.br   = (const float *)d_in[2];
  p.Wu   = (const float *)d_in[3];   p.bu   = (const float *)d_in[4];
  p.Wn   = (const float *)d_in[5];   p.bn   = (const float *)d_in[6];
  p.Wmr  = (const float *)d_in[7];   p.bmr  = (const float *)d_in[8];
  p.Wmw  = (const float *)d_in[9];   p.bmw  = (const float *)d_in[10];
  p.Wmu  = (const float *)d_in[11];  p.bmu  = (const float *)d_in[12];
  p.Wmh  = (const float *)d_in[13];  p.bmh  = (const float *)d_in[14];
  p.Wd   = (const float *)d_in[15];  p.bd   = (const float *)d_in[16];
  p.Wsr  = (const float *)d_in[17];  p.bs   = (const float *)d_in[18];
  p.Wout = (const float *)d_in[19];  p.bout = (const float *)d_in[20];
  p.dopap = (const float *)d_in[21];
  p.serop = (const float *)d_in[22];
  p.out  = (float *)d_out;

  char *w = (char *)d_ws;
  auto take = [&](size_t bytes) -> char * {
    char *r = w;
    w += (bytes + 255) & ~(size_t)255;
    return r;
  };
  p.bar  = (unsigned *)take(256);
  p.wr   = (bf16 *)take((size_t)Lc * Hc * IH * 2);
  p.wu   = (bf16 *)take((size_t)Lc * Hc * IH * 2);
  p.wn   = (bf16 *)take((size_t)Lc * Hc * IH * 2);
  p.wmr  = (bf16 *)take((size_t)Lc * Pc * Hc * 2);
  p.wmw  = (bf16 *)take((size_t)Lc * Pc * Hc * 2);
  p.wmu  = (bf16 *)take((size_t)Lc * Pc * Hc * 2);
  p.wd   = (bf16 *)take((size_t)Lc * Pc * Hc * 2);
  p.wsr  = (bf16 *)take((size_t)Lc * Pc * Hc * 2);
  p.wmh  = (bf16 *)take((size_t)Lc * Hc * Pc * 2);
  p.wout = (bf16 *)take((size_t)Oc * Hc * 2);
  p.xbf  = (bf16 *)take((size_t)Bc * Ic * 2);
  p.hbf  = (bf16 *)take((size_t)Lc * Bc * Hc * 2);
  p.rhbf = (bf16 *)take((size_t)Bc * Hc * 2);
  p.hcbf = (bf16 *)take((size_t)Bc * Hc * 2);
  p.mobf = (bf16 *)take((size_t)Bc * Pc * 2);
  p.h32  = (float *)take((size_t)Lc * Bc * Hc * 4);
  p.m32  = (float *)take((size_t)Lc * Bc * Pc * 4);
  p.u32  = (float *)take((size_t)Bc * Hc * 4);
  p.hc32 = (float *)take((size_t)Bc * Hc * 4);
  p.et32 = (float *)take((size_t)Bc * 1024 * 4);

  hipMemsetAsync(p.bar, 0, 256, stream);
  hipLaunchKernelGGL(gru_persistent, dim3(NWG), dim3(TPB), 0, stream, p);
}
